// MambaBlock_21663815041549
// MI455X (gfx1250) — matrix operand order repaired and checked by reading, not compiled, not run on hardware
//
#include <hip/hip_runtime.h>
#include <math.h>

// ---------------------------------------------------------------------------
// Problem sizes (compile-time constants from the reference)
// ---------------------------------------------------------------------------
#define BB      4
#define LL      1024
#define DM      1024           // d_model
#define DS      16             // d_state
#define ROWS    (BB * LL)      // 4096 rows of the flattened [B*L, D] matrices

#define KC      128            // K-chunk staged in LDS per iteration
#define KCP     (KC + 8)       // padded row stride (bf16 elems) -> bank-conflict free
#define NCHUNK  (DM / KC)      // 8

typedef __attribute__((ext_vector_type(16))) __bf16 v16bf;
typedef __attribute__((ext_vector_type(8)))  float  v8f;

union FragBF {            // 16 bf16 = 32 bytes, filled by two b128 loads
    v16bf v;
    uint4 q[2];
};

static __device__ __forceinline__ unsigned short f32_to_bf16(float f) {
    unsigned int u = __float_as_uint(f);
    u += 0x7FFFu + ((u >> 16) & 1u);       // round-to-nearest-even
    return (unsigned short)(u >> 16);
}

// CDNA5 async global->LDS copy (ASYNCcnt-tracked).  VDST = LDS byte offset,
// VADDR = 64-bit global address, saddr = off (null).
// IMPORTANT: the LDS pointer itself is passed as an operand so it escapes into
// the asm -- otherwise clang proves the __shared__ array is never written and
// folds the subsequent ds_load reads to undef.
static __device__ __forceinline__ void async_copy_b128(unsigned short* lds_ptr,
                                                       const void* gaddr) {
    const unsigned off = (unsigned)(unsigned long long)lds_ptr;  // low 32 bits = LDS offset
    asm volatile("global_load_async_to_lds_b128 %0, %1, off"
                 :
                 : "v"(off), "v"(gaddr), "v"(lds_ptr)
                 : "memory");
}
template <int N>
static __device__ __forceinline__ void wait_asynccnt() {
    asm volatile("s_wait_asynccnt %0" :: "i"(N) : "memory");
}

// ---------------------------------------------------------------------------
// 1) LayerNorm: one block per row (B*L rows).  Writes fp32 xn and bf16 xn.
// ---------------------------------------------------------------------------
__global__ void ln_kernel(const float* __restrict__ x,
                          const float* __restrict__ gamma,
                          const float* __restrict__ beta,
                          float* __restrict__ xn,
                          unsigned short* __restrict__ xnbf) {
    const int row  = blockIdx.x;
    const float* xr = x + (size_t)row * DM;

    float s = 0.f, ss = 0.f;
    for (int i = threadIdx.x; i < DM; i += 256) {
        float v = xr[i];
        s += v; ss += v * v;
    }
    for (int m = 16; m >= 1; m >>= 1) {
        s  += __shfl_xor(s,  m, 32);
        ss += __shfl_xor(ss, m, 32);
    }
    __shared__ float sh1[8], sh2[8];
    const int wave = threadIdx.x >> 5, lane = threadIdx.x & 31;
    if (lane == 0) { sh1[wave] = s; sh2[wave] = ss; }
    __syncthreads();
    if (wave == 0) {
        float a  = (lane < 8) ? sh1[lane] : 0.f;
        float b2 = (lane < 8) ? sh2[lane] : 0.f;
        for (int m = 4; m >= 1; m >>= 1) {
            a  += __shfl_xor(a,  m, 32);
            b2 += __shfl_xor(b2, m, 32);
        }
        if (lane == 0) { sh1[0] = a; sh2[0] = b2; }
    }
    __syncthreads();
    const float mean = sh1[0] * (1.f / DM);
    const float var  = sh2[0] * (1.f / DM) - mean * mean;
    const float rstd = rsqrtf(var + 1e-5f);

    for (int i = threadIdx.x; i < DM; i += 256) {
        float v = (xr[i] - mean) * rstd * gamma[i] + beta[i];
        xn  [(size_t)row * DM + i] = v;
        xnbf[(size_t)row * DM + i] = f32_to_bf16(v);
    }
}

// ---------------------------------------------------------------------------
// 2) Transpose + bf16 convert of W_delta:  WdT[n][k] = bf16(W[k][n])
// ---------------------------------------------------------------------------
__global__ void transpose_wd_kernel(const float* __restrict__ W,
                                    unsigned short* __restrict__ WT) {
    int id = blockIdx.x * 256 + threadIdx.x;       // over DM*DM
    int k = id & (DM - 1);
    int n = id >> 10;
    WT[(size_t)n * DM + k] = f32_to_bf16(W[(size_t)k * DM + n]);
}

// W_B / W_C are [DM][DS] -> WT [DS][DM]
__global__ void transpose_wbc_kernel(const float* __restrict__ WB,
                                     const float* __restrict__ WC,
                                     unsigned short* __restrict__ WBT,
                                     unsigned short* __restrict__ WCT) {
    const float* W          = blockIdx.y ? WC  : WB;
    unsigned short* WT      = blockIdx.y ? WCT : WBT;
    int id = blockIdx.x * 256 + threadIdx.x;       // over DM*DS
    int n = id & (DS - 1);
    int k = id >> 4;
    WT[(size_t)n * DM + k] = f32_to_bf16(W[(size_t)k * DS + n]);
}

// ---------------------------------------------------------------------------
// 3) WMMA GEMM:  delta = softplus(xn @ W_delta + b_delta)
//    Block: 256 threads = 8 waves, computes 256(M) x 64(N).
//    Wave: 32(M) x 64(N) tile -> 8 accumulators; 8 wmma per K-step.
//    B panel (64 cols x KC) double-buffered in LDS via async global->LDS
//    loads (ASYNCcnt) and consumed through ds_load_b128 (padded stride ->
//    bank-conflict free).  grid = (DM/64, ROWS/256) = (16, 16).
// ---------------------------------------------------------------------------
__global__ void gemm_delta_kernel(const unsigned short* __restrict__ Abf,  // [ROWS][DM] bf16 xn
                                  const unsigned short* __restrict__ WT,   // [DM][DM]   bf16 W^T
                                  const float* __restrict__ bias,          // [DM]
                                  float* __restrict__ out) {               // [ROWS][DM]
    __shared__ unsigned short Bs[2][64 * KCP];     // ~34 KB, double-buffered

    const int tid  = threadIdx.x;
    const int lane = tid & 31;
    const int wave = tid >> 5;
    const int half = lane >> 4;          // 0 or 1
    const int r    = lane & 15;
    const int m0   = blockIdx.y * 256 + wave * 32; // two 16-row tiles
    const int n0   = blockIdx.x * 64;

    v8f acc[2][4];
#pragma unroll
    for (int mt = 0; mt < 2; ++mt)
#pragma unroll
        for (int j = 0; j < 4; ++j) acc[mt][j] = (v8f){0,0,0,0,0,0,0,0};

    const unsigned short* arow0 = Abf + (size_t)(m0 + r) * DM;
    const unsigned short* arow1 = Abf + (size_t)(m0 + 16 + r) * DM;

    // cooperative async staging: thread -> (col = tid>>4 (+16/pass), v = tid&15)
    const int ld_v  = tid & 15;          // which 8-elem (16B) piece of the row
    const int ld_c0 = tid >> 4;          // starting column (0..15)

    // prologue: stage chunk 0 into buffer 0
#pragma unroll
    for (int pass = 0; pass < 4; ++pass) {
        const int col = ld_c0 + pass * 16;
        async_copy_b128(&Bs[0][col * KCP + ld_v * 8],
                        WT + (size_t)(n0 + col) * DM + ld_v * 8);
    }

    for (int c = 0; c < NCHUNK; ++c) {
        const int k0  = c * KC;
        const int buf = c & 1;

        if (c + 1 < NCHUNK) {
            // issue next chunk's async loads into the other buffer; they fly
            // under this chunk's 32 WMMAs.  (Buffer was last read in chunk
            // c-1 and released by the trailing barrier of that iteration.)
#pragma unroll
            for (int pass = 0; pass < 4; ++pass) {
                const int col = ld_c0 + pass * 16;
                async_copy_b128(&Bs[buf ^ 1][col * KCP + ld_v * 8],
                                WT + (size_t)(n0 + col) * DM + (k0 + KC) + ld_v * 8);
            }
            wait_asynccnt<4>();   // oldest 4 (= this chunk's loads) done; in-order
        } else {
            wait_asynccnt<0>();
        }
        __syncthreads();

#pragma unroll
        for (int kk = 0; kk < KC; kk += 32) {
            FragBF a0, a1;
            // A 16x32 bf16: half 0 -> K chunks [0..7],[16..23]; half 1 -> [8..15],[24..31]
            a0.q[0] = *(const uint4*)(arow0 + k0 + kk +      half * 8);
            a0.q[1] = *(const uint4*)(arow0 + k0 + kk + 16 + half * 8);
            a1.q[0] = *(const uint4*)(arow1 + k0 + kk +      half * 8);
            a1.q[1] = *(const uint4*)(arow1 + k0 + kk + 16 + half * 8);
#pragma unroll
            for (int j = 0; j < 4; ++j) {
                // B 32x16 bf16: half 0 -> K [0..7],[16..23]; half 1 -> [8..15],[24..31], N = r
                const unsigned short* bp = &Bs[buf][(j * 16 + r) * KCP + kk + half * 8];
                FragBF b;
                b.q[0] = *(const uint4*)(bp);        // ds_load_b128
                b.q[1] = *(const uint4*)(bp + 16);   // ds_load_b128
                acc[0][j] = __builtin_amdgcn_wmma_f32_16x16x32_bf16(
                    false, a0.v, false, b.v, (short)0, acc[0][j], false, false);
                acc[1][j] = __builtin_amdgcn_wmma_f32_16x16x32_bf16(
                    false, a1.v, false, b.v, (short)0, acc[1][j], false, false);
            }
        }
        __syncthreads();   // all waves done reading Bs[buf] before it is re-staged
    }

#pragma unroll
    for (int mt = 0; mt < 2; ++mt) {
#pragma unroll
        for (int j = 0; j < 4; ++j) {
            const int col = n0 + j * 16 + r;
            const float bv = bias[col];
#pragma unroll
            for (int e = 0; e < 8; ++e) {
                const int row = m0 + mt * 16 + half * 8 + e;  // C/D: M = half*8+e, N = r
                float v = acc[mt][j][e] + bv;
                float sp = (v > 20.0f) ? v : __logf(1.0f + __expf(v));  // softplus
                out[(size_t)row * DM + col] = sp;
            }
        }
    }
}

// ---------------------------------------------------------------------------
// 4) WMMA GEMM for Bm and Cm (N = 16 -> exactly one tile wide).
//    grid = (ROWS/128, 2); blockIdx.y selects B vs C.
// ---------------------------------------------------------------------------
__global__ void gemm_bc_kernel(const unsigned short* __restrict__ Abf,
                               const unsigned short* __restrict__ WBT,
                               const unsigned short* __restrict__ WCT,
                               const float* __restrict__ bB,
                               const float* __restrict__ bC,
                               float* __restrict__ Bm,
                               float* __restrict__ Cm) {
    const unsigned short* WT = blockIdx.y ? WCT : WBT;
    const float* bias        = blockIdx.y ? bC  : bB;
    float* out               = blockIdx.y ? Cm  : Bm;

    const int lane = threadIdx.x & 31;
    const int wave = threadIdx.x >> 5;
    const int half = lane >> 4;
    const int r    = lane & 15;
    const int m0   = blockIdx.x * 128 + wave * 16;

    v8f acc = (v8f){0,0,0,0,0,0,0,0};
    const unsigned short* arow  = Abf + (size_t)(m0 + r) * DM;
    const unsigned short* brow0 = WT + (size_t)r * DM;

    for (int k0 = 0; k0 < DM; k0 += 32) {
        FragBF a, b;
        a.q[0] = *(const uint4*)(arow + k0 +      half * 8);
        a.q[1] = *(const uint4*)(arow + k0 + 16 + half * 8);
        b.q[0] = *(const uint4*)(brow0 + k0 +      half * 8);
        b.q[1] = *(const uint4*)(brow0 + k0 + 16 + half * 8);
        acc = __builtin_amdgcn_wmma_f32_16x16x32_bf16(
            false, a.v, false, b.v, (short)0, acc, false, false);
    }

    const float bv = bias[r];
#pragma unroll
    for (int e = 0; e < 8; ++e) {
        const int row = m0 + half * 8 + e;
        out[(size_t)row * DS + r] = acc[e] + bv;
    }
}

// ---------------------------------------------------------------------------
// 5) Selective scan.  One lane per (b, d, n) state element; h lives in a
//    register for all L steps.  Wave32 = two (b,d) channels per wave; y is a
//    16-lane xor-shuffle reduction over n.
// ---------------------------------------------------------------------------
__global__ void scan_kernel(const float* __restrict__ x,      // residual [B,L,D]
                            const float* __restrict__ xn,     // [B,L,D]
                            const float* __restrict__ delta,  // [B,L,D]
                            const float* __restrict__ Bm,     // [B,L,N]
                            const float* __restrict__ Cm,     // [B,L,N]
                            const float* __restrict__ A_log,  // [D,N]
                            const float* __restrict__ D_skip, // [D]
                            float* __restrict__ out) {        // [B,L,D]
    const int tid  = blockIdx.x * 256 + threadIdx.x;
    const int lane = tid & 31;
    const int half = lane >> 4;
    const int n    = lane & 15;
    const int ch   = (tid >> 5) * 2 + half;     // 0 .. B*D-1
    const int b    = ch >> 10;
    const int d    = ch & (DM - 1);

    const float A  = -__expf(A_log[d * DS + n]);
    const float ds = D_skip[d];
    float h = 0.f;

    for (int t = 0; t < LL; ++t) {
        const size_t row = (size_t)(b * LL + t);
        const float dt = delta[row * DM + d];
        const float xt = xn  [row * DM + d];
        const float Bn = Bm  [row * DS + n];
        const float Cn = Cm  [row * DS + n];

        if (t + 1 < LL) {   // global_prefetch_b8 for next timestep
            __builtin_prefetch(delta + (row + 1) * DM + d, 0, 1);
            __builtin_prefetch(xn    + (row + 1) * DM + d, 0, 1);
        }

        const float dA = __expf(dt * A);
        h = dA * h + (dt * xt) * Bn;

        float yv = h * Cn;                       // reduce over n within half-wave
        yv += __shfl_xor(yv, 1, 16);
        yv += __shfl_xor(yv, 2, 16);
        yv += __shfl_xor(yv, 4, 16);
        yv += __shfl_xor(yv, 8, 16);

        if (n == 0)
            out[row * DM + d] = x[row * DM + d] + ds * xt + yv;
    }
}

// ---------------------------------------------------------------------------
// Launcher.  Workspace ~42.6 MB, laid out below.
// ---------------------------------------------------------------------------
extern "C" void kernel_launch(void* const* d_in, const int* in_sizes, int n_in,
                              void* d_out, int out_size, void* d_ws, size_t ws_size,
                              hipStream_t stream) {
    (void)in_sizes; (void)n_in; (void)out_size; (void)ws_size;

    const float* x       = (const float*)d_in[0];
    const float* gamma   = (const float*)d_in[1];
    const float* beta    = (const float*)d_in[2];
    const float* W_delta = (const float*)d_in[3];
    const float* b_delta = (const float*)d_in[4];
    const float* W_B     = (const float*)d_in[5];
    const float* b_B     = (const float*)d_in[6];
    const float* W_C     = (const float*)d_in[7];
    const float* b_C     = (const float*)d_in[8];
    const float* A_log   = (const float*)d_in[9];
    const float* D_skip  = (const float*)d_in[10];
    float* out = (float*)d_out;

    char* ws = (char*)d_ws;
    size_t off = 0;
    float*          xn    = (float*)(ws + off);          off += (size_t)ROWS * DM * 4;
    unsigned short* xnbf  = (unsigned short*)(ws + off); off += (size_t)ROWS * DM * 2;
    unsigned short* WdT   = (unsigned short*)(ws + off); off += (size_t)DM * DM * 2;
    unsigned short* WBT   = (unsigned short*)(ws + off); off += (size_t)DS * DM * 2;
    unsigned short* WCT   = (unsigned short*)(ws + off); off += (size_t)DS * DM * 2;
    float*          delta = (float*)(ws + off);          off += (size_t)ROWS * DM * 4;
    float*          Bm    = (float*)(ws + off);          off += (size_t)ROWS * DS * 4;
    float*          Cm    = (float*)(ws + off);          off += (size_t)ROWS * DS * 4;

    ln_kernel<<<ROWS, 256, 0, stream>>>(x, gamma, beta, xn, xnbf);

    transpose_wd_kernel<<<(DM * DM) / 256, 256, 0, stream>>>(W_delta, WdT);
    transpose_wbc_kernel<<<dim3((DM * DS) / 256, 2), 256, 0, stream>>>(W_B, W_C, WBT, WCT);

    gemm_delta_kernel<<<dim3(DM / 64, ROWS / 256), 256, 0, stream>>>(xnbf, WdT, b_delta, delta);
    gemm_bc_kernel<<<dim3(ROWS / 128, 2), 256, 0, stream>>>(xnbf, WBT, WCT, b_B, b_C, Bm, Cm);

    scan_kernel<<<(BB * DM * DS) / 256, 256, 0, stream>>>(
        x, xn, delta, Bm, Cm, A_log, D_skip, out);
}
